// EncoderLSTMContinuous_47571057770805
// MI455X (gfx1250) — matrix operand order repaired and checked by reading, not compiled, not run on hardware
//
#include <hip/hip_runtime.h>
#include <hip/hip_bf16.h>
#include <math.h>

#define N_ROWS 8192
#define IN_DIM 1024
#define UNITS  512
#define ZDEPTH 32
#define ZK     256

#define RB   32    // rows per block in the scan kernel (2 row-tiles of 16)
#define LDAH 520   // padded fp32 row stride for h_prev tile (bank-spread)
#define LDAB 520   // padded bf16 row stride (multiple of 8 -> 16B aligned rows)

typedef __bf16 bf16_t;
typedef bf16_t v16bf __attribute__((ext_vector_type(16)));
typedef bf16_t v8bf  __attribute__((ext_vector_type(8)));
typedef float  v8f   __attribute__((ext_vector_type(8)));

__device__ __forceinline__ v16bf mk16(v8bf lo, v8bf hi) {
    return __builtin_shufflevector(lo, hi, 0,1,2,3,4,5,6,7,8,9,10,11,12,13,14,15);
}

// A-matrix fragment (16x32 bf16, MxK) from an LDS tile with row stride LDAB.
// lanes 0-15 -> M=lane&15; half=lane>>4:
// elems 0..7 = A[m][k0+half*8+e], elems 8..15 = A[m][k0+16+half*8+e]
__device__ __forceinline__ v16bf load_a_lds(const bf16_t* base, int row0, int k0, int lane) {
    const int m = lane & 15, half = lane >> 4;
    const bf16_t* p = base + (row0 + m) * LDAB + k0 + half * 8;
    v8bf lo = *(const v8bf*)p;          // 16B aligned
    v8bf hi = *(const v8bf*)(p + 16);
    return mk16(lo, hi);
}

// B-matrix fragment (32x16 bf16, KxN) from transposed weight Wt[n][k] in global.
// n = lane&15; elems 0..7 = Wt[n][k0+half*8+e], elems 8..15 = Wt[n][k0+16+half*8+e]
__device__ __forceinline__ v16bf load_b_glb(const bf16_t* __restrict__ wt, int ldw,
                                            int col0, int k0, int lane) {
    const int n = lane & 15, half = lane >> 4;
    const bf16_t* p = wt + (size_t)(col0 + n) * ldw + k0 + half * 8;
    v8bf lo = *(const v8bf*)p;
    v8bf hi = *(const v8bf*)(p + 16);
    return mk16(lo, hi);
}

#define WMMA_BF16(a, b, c) \
    __builtin_amdgcn_wmma_f32_16x16x32_bf16(false, (a), false, (b), (short)0, (c), false, false)

__device__ __forceinline__ float sigm(float x) { return 1.0f / (1.0f + __expf(-x)); }

// ---------------------------------------------------------------------------
__global__ __launch_bounds__(256)
void cvt_bf16_kernel(const float* __restrict__ src, bf16_t* __restrict__ dst, int n) {
    for (int i = blockIdx.x * blockDim.x + threadIdx.x; i < n; i += gridDim.x * blockDim.x)
        dst[i] = (bf16_t)src[i];
}

// fp32 W[K][Nc] -> bf16 Wt[Nc][K] (transpose + convert)
__global__ __launch_bounds__(256)
void transpose_cvt_kernel(const float* __restrict__ src, bf16_t* __restrict__ dst,
                          int K, int Nc) {
    int total = K * Nc;
    for (int idx = blockIdx.x * blockDim.x + threadIdx.x; idx < total;
         idx += gridDim.x * blockDim.x) {
        int n = idx / K, k = idx - n * K;
        dst[(size_t)n * K + k] = (bf16_t)src[(size_t)k * Nc + n];
    }
}

// ---------------------------------------------------------------------------
// xh = x @ h_W + h_b : [8192x1024] @ [1024x512] -> fp32 [8192x512]
// Each wave computes a 32x16 output strip (2 row-tiles) sharing one B-fragment.
__global__ __launch_bounds__(256)
void xh_gemm_kernel(const bf16_t* __restrict__ xbf, const bf16_t* __restrict__ hWt,
                    const float* __restrict__ h_b, float* __restrict__ xh) {
    const int lane = threadIdx.x & 31;
    const int w    = threadIdx.x >> 5;
    const int job  = blockIdx.x * 8 + w;          // 8192 jobs (256 row-pairs x 32 col)
    const int rp = job >> 5;                      // row-pair 0..255 (32 rows each)
    const int ct = job & 31;                      // col tile 0..31
    const int n = lane & 15, half = lane >> 4;
    const int r0 = rp * 32;

    v8f acc0 = {}, acc1 = {};
    const bf16_t* arow0 = xbf + (size_t)(r0 + n) * IN_DIM;
    const bf16_t* arow1 = xbf + (size_t)(r0 + 16 + n) * IN_DIM;
    const bf16_t* brow  = hWt + (size_t)(ct * 16 + n) * IN_DIM;
    for (int kc = 0; kc < IN_DIM / 32; ++kc) {
        const int k0 = kc << 5;
        const bf16_t* bp = brow + k0 + half * 8;
        v16bf b = mk16(*(const v8bf*)bp, *(const v8bf*)(bp + 16));
        const bf16_t* a0p = arow0 + k0 + half * 8;
        const bf16_t* a1p = arow1 + k0 + half * 8;
        v16bf a0 = mk16(*(const v8bf*)a0p, *(const v8bf*)(a0p + 16));
        v16bf a1 = mk16(*(const v8bf*)a1p, *(const v8bf*)(a1p + 16));
        acc0 = WMMA_BF16(a0, b, acc0);
        acc1 = WMMA_BF16(a1, b, acc1);
    }
    const int col = ct * 16 + n;
    const float bias = h_b[col];
    for (int r = 0; r < 8; ++r) {
        const int m = half * 8 + r;               // D layout: M = half*8 + r
        xh[(size_t)(r0 + m) * UNITS + col]      = acc0[r] + bias;
        xh[(size_t)(r0 + 16 + m) * UNITS + col] = acc1[r] + bias;
    }
}

// ---------------------------------------------------------------------------
// The scan: each block owns RB=32 rows for all 32 steps. h_prev lives in LDS.
// All GEMM phases register-block both row-tiles against one B-fragment load.
__global__ __launch_bounds__(256)
void lstm_scan_kernel(const float* __restrict__ xh,
                      const bf16_t* __restrict__ hUt, const bf16_t* __restrict__ fWt,
                      const bf16_t* __restrict__ iWt, const bf16_t* __restrict__ cWt,
                      const bf16_t* __restrict__ oWt, const bf16_t* __restrict__ tWt,
                      const bf16_t* __restrict__ yWt,
                      const float* __restrict__ f_b, const float* __restrict__ i_b,
                      const float* __restrict__ c_b, const float* __restrict__ o_b,
                      const float* __restrict__ t_b, const float* __restrict__ y_b,
                      const float* __restrict__ h0, float* __restrict__ out) {
    __shared__ float  s_h[RB][LDAH];       // fp32 h_prev tile (persistent)
    __shared__ bf16_t s_a[2][RB][LDAB];    // ping-pong bf16 A operands

    const int tid  = threadIdx.x;
    const int lane = tid & 31;
    const int w    = tid >> 5;
    const int row0 = blockIdx.x * RB;
    const int n    = lane & 15, half = lane >> 4;

    // init h_prev from broadcast h0
    for (int idx = tid; idx < RB * UNITS; idx += 256)
        s_h[idx >> 9][idx & 511] = h0[idx & 511];
    __syncthreads();

    for (int s = 0; s < ZDEPTH; ++s) {
        // --- phase A: bf16 copy of h_prev into A-buffer 0 ---
        for (int idx = tid; idx < RB * UNITS; idx += 256) {
            const int r = idx >> 9, u = idx & 511;
            s_a[0][r][u] = (bf16_t)s_h[r][u];
        }
        __syncthreads();

        // --- phase B: h = tanh(xh + h_prev @ h_U)  -> bf16 in A-buffer 1 ---
        for (int j = 0; j < 4; ++j) {              // 32 col tiles / 8 waves
            const int ct = w + (j << 3);
            v8f acc0 = {}, acc1 = {};
            for (int kc = 0; kc < UNITS / 32; ++kc) {
                const int k0 = kc << 5;
                v16bf b  = load_b_glb(hUt, UNITS, ct * 16, k0, lane);
                v16bf a0 = load_a_lds(&s_a[0][0][0], 0,  k0, lane);
                v16bf a1 = load_a_lds(&s_a[0][0][0], 16, k0, lane);
                acc0 = WMMA_BF16(a0, b, acc0);
                acc1 = WMMA_BF16(a1, b, acc1);
            }
            const int col = ct * 16 + n;
            for (int r = 0; r < 8; ++r) {
                const int m0 = half * 8 + r, m1 = 16 + half * 8 + r;
                s_a[1][m0][col] =
                    (bf16_t)tanhf(acc0[r] + xh[(size_t)(row0 + m0) * UNITS + col]);
                s_a[1][m1][col] =
                    (bf16_t)tanhf(acc1[r] + xh[(size_t)(row0 + m1) * UNITS + col]);
            }
        }
        __syncthreads();

        // --- phase C: fused gates f,i,c,o; h1 = h_prev*f + c*i; u = o*h1 ---
        for (int j = 0; j < 4; ++j) {
            const int ct = w + (j << 3);
            v8f f0 = {}, f1 = {}, i0 = {}, i1 = {}, c0 = {}, c1 = {}, o0 = {}, o1 = {};
            for (int kc = 0; kc < UNITS / 32; ++kc) {
                const int k0 = kc << 5;
                v16bf a0 = load_a_lds(&s_a[1][0][0], 0,  k0, lane);
                v16bf a1 = load_a_lds(&s_a[1][0][0], 16, k0, lane);
                v16bf bF = load_b_glb(fWt, UNITS, ct * 16, k0, lane);
                f0 = WMMA_BF16(a0, bF, f0);  f1 = WMMA_BF16(a1, bF, f1);
                v16bf bI = load_b_glb(iWt, UNITS, ct * 16, k0, lane);
                i0 = WMMA_BF16(a0, bI, i0);  i1 = WMMA_BF16(a1, bI, i1);
                v16bf bC = load_b_glb(cWt, UNITS, ct * 16, k0, lane);
                c0 = WMMA_BF16(a0, bC, c0);  c1 = WMMA_BF16(a1, bC, c1);
                v16bf bO = load_b_glb(oWt, UNITS, ct * 16, k0, lane);
                o0 = WMMA_BF16(a0, bO, o0);  o1 = WMMA_BF16(a1, bO, o1);
            }
            const int col = ct * 16 + n;
            const float fb = f_b[col], ib = i_b[col], cb = c_b[col], ob = o_b[col];
            for (int r = 0; r < 8; ++r) {
                {   // row-tile 0
                    const int rl = half * 8 + r;
                    float f  = sigm(f0[r] + fb);
                    float ii = sigm(i0[r] + ib);
                    float cc = tanhf(c0[r] + cb);
                    float o  = sigm(o0[r] + ob);
                    float h1 = s_h[rl][col] * f + cc * ii;
                    s_h[rl][col]    = h1;
                    s_a[0][rl][col] = (bf16_t)(o * h1);
                }
                {   // row-tile 1
                    const int rl = 16 + half * 8 + r;
                    float f  = sigm(f1[r] + fb);
                    float ii = sigm(i1[r] + ib);
                    float cc = tanhf(c1[r] + cb);
                    float o  = sigm(o1[r] + ob);
                    float h1 = s_h[rl][col] * f + cc * ii;
                    s_h[rl][col]    = h1;
                    s_a[0][rl][col] = (bf16_t)(o * h1);
                }
            }
        }
        __syncthreads();

        // --- phase D: t = tanh(u @ t_W + t_b) -> bf16 in A-buffer 1 ---
        for (int j = 0; j < 4; ++j) {
            const int ct = w + (j << 3);
            v8f acc0 = {}, acc1 = {};
            for (int kc = 0; kc < UNITS / 32; ++kc) {
                const int k0 = kc << 5;
                v16bf b  = load_b_glb(tWt, UNITS, ct * 16, k0, lane);
                v16bf a0 = load_a_lds(&s_a[0][0][0], 0,  k0, lane);
                v16bf a1 = load_a_lds(&s_a[0][0][0], 16, k0, lane);
                acc0 = WMMA_BF16(a0, b, acc0);
                acc1 = WMMA_BF16(a1, b, acc1);
            }
            const int col = ct * 16 + n;
            const float tb = t_b[col];
            for (int r = 0; r < 8; ++r) {
                const int m0 = half * 8 + r, m1 = 16 + half * 8 + r;
                s_a[1][m0][col] = (bf16_t)tanhf(acc0[r] + tb);
                s_a[1][m1][col] = (bf16_t)tanhf(acc1[r] + tb);
            }
        }
        __syncthreads();

        // --- phase E: y = tanh(t @ y_W + y_b) -> global [N][ZDEPTH][ZK] ---
        for (int j = 0; j < 2; ++j) {              // 16 col tiles / 8 waves
            const int ct = w + (j << 3);
            v8f acc0 = {}, acc1 = {};
            for (int kc = 0; kc < UNITS / 32; ++kc) {
                const int k0 = kc << 5;
                v16bf b  = load_b_glb(yWt, UNITS, ct * 16, k0, lane);
                v16bf a0 = load_a_lds(&s_a[1][0][0], 0,  k0, lane);
                v16bf a1 = load_a_lds(&s_a[1][0][0], 16, k0, lane);
                acc0 = WMMA_BF16(a0, b, acc0);
                acc1 = WMMA_BF16(a1, b, acc1);
            }
            const int col = ct * 16 + n;
            const float yb = y_b[col];
            for (int r = 0; r < 8; ++r) {
                const int m0 = half * 8 + r, m1 = 16 + half * 8 + r;
                out[(size_t)(row0 + m0) * (ZDEPTH * ZK) + (size_t)s * ZK + col] =
                    tanhf(acc0[r] + yb);
                out[(size_t)(row0 + m1) * (ZDEPTH * ZK) + (size_t)s * ZK + col] =
                    tanhf(acc1[r] + yb);
            }
        }
        __syncthreads();
    }
}

// ---------------------------------------------------------------------------
extern "C" void kernel_launch(void* const* d_in, const int* in_sizes, int n_in,
                              void* d_out, int out_size, void* d_ws, size_t ws_size,
                              hipStream_t stream) {
    const float* x   = (const float*)d_in[0];
    const float* h_W = (const float*)d_in[1];
    const float* h_U = (const float*)d_in[2];
    const float* h_b = (const float*)d_in[3];
    const float* f_W = (const float*)d_in[4];
    const float* f_b = (const float*)d_in[5];
    const float* i_W = (const float*)d_in[6];
    const float* i_b = (const float*)d_in[7];
    const float* c_W = (const float*)d_in[8];
    const float* c_b = (const float*)d_in[9];
    const float* o_W = (const float*)d_in[10];
    const float* o_b = (const float*)d_in[11];
    const float* t_W = (const float*)d_in[12];
    const float* t_b = (const float*)d_in[13];
    const float* y_W = (const float*)d_in[14];
    const float* y_b = (const float*)d_in[15];
    const float* h0  = (const float*)d_in[16];
    float* out = (float*)d_out;

    char* ws = (char*)d_ws;
    bf16_t* xbf = (bf16_t*)(ws + 0);          // 8192*1024*2 = 16 MB
    bf16_t* hWt = (bf16_t*)(ws + 16777216);   // 512*1024*2  = 1 MB
    bf16_t* hUt = (bf16_t*)(ws + 17825792);   // 512*512*2 each below
    bf16_t* fWt = (bf16_t*)(ws + 18350080);
    bf16_t* iWt = (bf16_t*)(ws + 18874368);
    bf16_t* cWt = (bf16_t*)(ws + 19398656);
    bf16_t* oWt = (bf16_t*)(ws + 19922944);
    bf16_t* tWt = (bf16_t*)(ws + 20447232);
    bf16_t* yWt = (bf16_t*)(ws + 20971520);   // 256*512*2
    float*  xh  = (float*)(ws + 21233664);    // 8192*512*4 = 16 MB

    cvt_bf16_kernel<<<2048, 256, 0, stream>>>(x, xbf, N_ROWS * IN_DIM);
    transpose_cvt_kernel<<<1024, 256, 0, stream>>>(h_W, hWt, IN_DIM, UNITS);
    transpose_cvt_kernel<<<512, 256, 0, stream>>>(h_U, hUt, UNITS, UNITS);
    transpose_cvt_kernel<<<512, 256, 0, stream>>>(f_W, fWt, UNITS, UNITS);
    transpose_cvt_kernel<<<512, 256, 0, stream>>>(i_W, iWt, UNITS, UNITS);
    transpose_cvt_kernel<<<512, 256, 0, stream>>>(c_W, cWt, UNITS, UNITS);
    transpose_cvt_kernel<<<512, 256, 0, stream>>>(o_W, oWt, UNITS, UNITS);
    transpose_cvt_kernel<<<512, 256, 0, stream>>>(t_W, tWt, UNITS, UNITS);
    transpose_cvt_kernel<<<256, 256, 0, stream>>>(y_W, yWt, UNITS, ZK);

    // xh = x @ h_W + h_b : 8192 wave-jobs (2 row-tiles each), 8 waves/block
    xh_gemm_kernel<<<1024, 256, 0, stream>>>(xbf, hWt, h_b, xh);

    // the scan: 8192/32 = 256 blocks, each carries its rows through all steps
    lstm_scan_kernel<<<N_ROWS / RB, 256, 0, stream>>>(
        xh, hUt, fWt, iWt, cWt, oWt, tWt, yWt,
        f_b, i_b, c_b, o_b, t_b, y_b, h0, out);
}
